// encoder_block_48722109006374
// MI455X (gfx1250) — compile-verified, matrix operand order checked by reading, not run on hardware
//
#include <hip/hip_runtime.h>
#include <hip/hip_bf16.h>
#include <math.h>

// ---------------- problem constants ----------------
static constexpr int Bb  = 8;
static constexpr int Nn  = 1024;
static constexpr int Cc  = 768;
static constexpr int Hh  = 12;
static constexpr int Dd  = 64;
static constexpr int HID = 3072;
static constexpr float SCALE = 0.125f;          // D^-0.5
static constexpr float EPS   = 1e-5f;

typedef __bf16 bf16_t;
typedef __attribute__((ext_vector_type(16))) __bf16 bf16x16;
typedef __attribute__((ext_vector_type(8)))  __bf16 bf16x8;
typedef __attribute__((ext_vector_type(8)))  float  f32x8;
typedef __attribute__((ext_vector_type(4)))  float  f32x4;

// ---------------- workspace layout (bytes) ----------------
// region A (reused):  [xln | q | k | vT]  ==  hid  (exactly 4*12.58MB = 50.33MB)
static constexpr size_t SZ_ACT   = (size_t)Bb * Nn * Cc * 2;          // 12,582,912
static constexpr size_t OFF_XLN  = 0;
static constexpr size_t OFF_Q    = OFF_XLN + SZ_ACT;
static constexpr size_t OFF_K    = OFF_Q   + SZ_ACT;
static constexpr size_t OFF_VT   = OFF_K   + SZ_ACT;
static constexpr size_t OFF_HID  = 0;                                  // alias over xln/q/k/vT
static constexpr size_t OFF_AV   = OFF_VT  + SZ_ACT;                   // bf16 [B,N,C]
static constexpr size_t OFF_X1   = OFF_AV  + SZ_ACT;                   // fp32 [B,N,C]
static constexpr size_t OFF_H2   = OFF_X1  + (size_t)Bb*Nn*Cc*4;       // bf16 [B,N,C]
static constexpr size_t OFF_WQKV = OFF_H2  + SZ_ACT;
static constexpr size_t OFF_WPRJ = OFF_WQKV + (size_t)3*Cc*Cc*2;
static constexpr size_t OFF_WFC1 = OFF_WPRJ + (size_t)Cc*Cc*2;
static constexpr size_t OFF_WFC2 = OFF_WFC1 + (size_t)HID*Cc*2;

// ---------------- WMMA helpers ----------------
__device__ __forceinline__ f32x8 wmma_bf16(bf16x16 a, bf16x16 b, f32x8 c) {
  return __builtin_amdgcn_wmma_f32_16x16x32_bf16(false, a, false, b, (short)0, c, false, false);
}

// A/B fragment for 16x16x32 bf16: lane row = lane%16, two contiguous 8-elem
// chunks at k offsets {hi*8} and {16+hi*8}  (hi = lane/16).
__device__ __forceinline__ bf16x16 load_frag_bf16(const bf16_t* rowptr, int hi) {
  union { bf16x16 v; bf16x8 h[2]; } f;
  f.h[0] = *(const bf16x8*)(rowptr + hi * 8);
  f.h[1] = *(const bf16x8*)(rowptr + 16 + hi * 8);
  return f.v;
}

// Same fragment but converting fp32 memory -> bf16 on the fly.
__device__ __forceinline__ bf16x16 load_frag_f32(const float* rowptr, int hi) {
  const float* p0 = rowptr + hi * 8;
  const float* p1 = rowptr + 16 + hi * 8;
  bf16x16 v;
#pragma unroll
  for (int i = 0; i < 8; ++i) {
    v[i]     = (bf16_t)p0[i];
    v[i + 8] = (bf16_t)p1[i];
  }
  return v;
}

// ---------------- ping-pong fragment set (32x64 tile: 2 A frags, 4 B frags) ----------------
struct FragSet {
  bf16x16 a0, a1, b0, b1, b2, b3;
};

__device__ __forceinline__ void load_set_bf16(FragSet& f,
                                              const bf16_t* ar0, const bf16_t* ar1,
                                              const bf16_t* b0p, const bf16_t* b1p,
                                              const bf16_t* b2p, const bf16_t* b3p,
                                              int k0, int hi) {
  f.a0 = load_frag_bf16(ar0 + k0, hi);
  f.a1 = load_frag_bf16(ar1 + k0, hi);
  f.b0 = load_frag_bf16(b0p + k0, hi);
  f.b1 = load_frag_bf16(b1p + k0, hi);
  f.b2 = load_frag_bf16(b2p + k0, hi);
  f.b3 = load_frag_bf16(b3p + k0, hi);
}

__device__ __forceinline__ void load_set_f32a(FragSet& f,
                                              const float* ar0, const float* ar1,
                                              const bf16_t* b0p, const bf16_t* b1p,
                                              const bf16_t* b2p, const bf16_t* b3p,
                                              int k0, int hi) {
  f.a0 = load_frag_f32(ar0 + k0, hi);
  f.a1 = load_frag_f32(ar1 + k0, hi);
  f.b0 = load_frag_bf16(b0p + k0, hi);
  f.b1 = load_frag_bf16(b1p + k0, hi);
  f.b2 = load_frag_bf16(b2p + k0, hi);
  f.b3 = load_frag_bf16(b3p + k0, hi);
}

__device__ __forceinline__ void mma_set(const FragSet& f, f32x8 acc[2][4]) {
  acc[0][0] = wmma_bf16(f.a0, f.b0, acc[0][0]);
  acc[0][1] = wmma_bf16(f.a0, f.b1, acc[0][1]);
  acc[0][2] = wmma_bf16(f.a0, f.b2, acc[0][2]);
  acc[0][3] = wmma_bf16(f.a0, f.b3, acc[0][3]);
  acc[1][0] = wmma_bf16(f.a1, f.b0, acc[1][0]);
  acc[1][1] = wmma_bf16(f.a1, f.b1, acc[1][1]);
  acc[1][2] = wmma_bf16(f.a1, f.b2, acc[1][2]);
  acc[1][3] = wmma_bf16(f.a1, f.b3, acc[1][3]);
}

// ---------------- software-pipelined 32(M) x 64(N) x K GEMM cores ----------------
// Explicit double buffering (ping-pong register sets, k-loop unrolled by 2) so
// the rotation needs no v_mov copies; loads overlap the WMMA batches.
// Requires K % 64 == 0 (all our K: 768, 1024, 3072).
__device__ __forceinline__ void gemm_core_bf16(const bf16_t* __restrict__ arow0,
                                               const bf16_t* __restrict__ arow1,
                                               const bf16_t* __restrict__ b0p,
                                               const bf16_t* __restrict__ b1p,
                                               const bf16_t* __restrict__ b2p,
                                               const bf16_t* __restrict__ b3p,
                                               int K, int hi, f32x8 acc[2][4]) {
  FragSet f0, f1;
  load_set_bf16(f0, arow0, arow1, b0p, b1p, b2p, b3p, 0, hi);
  int k0 = 32;
  for (; k0 + 32 < K; k0 += 64) {
    load_set_bf16(f1, arow0, arow1, b0p, b1p, b2p, b3p, k0, hi);
    mma_set(f0, acc);
    load_set_bf16(f0, arow0, arow1, b0p, b1p, b2p, b3p, k0 + 32, hi);
    mma_set(f1, acc);
  }
  load_set_bf16(f1, arow0, arow1, b0p, b1p, b2p, b3p, k0, hi);   // k0 == K-32
  mma_set(f0, acc);
  mma_set(f1, acc);
}

__device__ __forceinline__ void gemm_core_f32a(const float* __restrict__ arow0,
                                               const float* __restrict__ arow1,
                                               const bf16_t* __restrict__ b0p,
                                               const bf16_t* __restrict__ b1p,
                                               const bf16_t* __restrict__ b2p,
                                               const bf16_t* __restrict__ b3p,
                                               int K, int hi, f32x8 acc[2][4]) {
  FragSet f0, f1;
  load_set_f32a(f0, arow0, arow1, b0p, b1p, b2p, b3p, 0, hi);
  int k0 = 32;
  for (; k0 + 32 < K; k0 += 64) {
    load_set_f32a(f1, arow0, arow1, b0p, b1p, b2p, b3p, k0, hi);
    mma_set(f0, acc);
    load_set_f32a(f0, arow0, arow1, b0p, b1p, b2p, b3p, k0 + 32, hi);
    mma_set(f1, acc);
  }
  load_set_f32a(f1, arow0, arow1, b0p, b1p, b2p, b3p, k0, hi);   // k0 == K-32
  mma_set(f0, acc);
  mma_set(f1, acc);
}

// ---------------- elementwise: fp32 -> bf16 weight convert ----------------
__global__ __launch_bounds__(256) void cvt_kernel(const float* __restrict__ in,
                                                  bf16_t* __restrict__ out, int n) {
  int i = blockIdx.x * 256 + threadIdx.x;
  if (i < n) out[i] = (bf16_t)in[i];
}

// ---------------- LayerNorm (row = 768), out bf16 ----------------
__global__ __launch_bounds__(256) void ln_kernel(const float* __restrict__ x,
                                                 const float* __restrict__ g,
                                                 const float* __restrict__ bta,
                                                 bf16_t* __restrict__ out) {
  __shared__ float red[256];
  __shared__ float s_mu, s_rstd;
  int row = blockIdx.x, tid = threadIdx.x;
  const float* xr = x + (size_t)row * Cc;
  float v[3];
#pragma unroll
  for (int i = 0; i < 3; ++i) v[i] = xr[tid + i * 256];
  float s = v[0] + v[1] + v[2];
  red[tid] = s; __syncthreads();
  for (int off = 128; off > 0; off >>= 1) {
    if (tid < off) red[tid] += red[tid + off];
    __syncthreads();
  }
  if (tid == 0) s_mu = red[0] * (1.0f / Cc);
  __syncthreads();
  float mu = s_mu;
  float s2 = 0.f;
#pragma unroll
  for (int i = 0; i < 3; ++i) { float d = v[i] - mu; s2 += d * d; }
  red[tid] = s2; __syncthreads();
  for (int off = 128; off > 0; off >>= 1) {
    if (tid < off) red[tid] += red[tid + off];
    __syncthreads();
  }
  if (tid == 0) s_rstd = rsqrtf(red[0] * (1.0f / Cc) + EPS);
  __syncthreads();
  float rstd = s_rstd;
#pragma unroll
  for (int i = 0; i < 3; ++i) {
    int c = tid + i * 256;
    out[(size_t)row * Cc + c] = (bf16_t)((v[i] - mu) * rstd * g[c] + bta[c]);
  }
}

// ---------------- QKV GEMM: [8192,768] x [2304,768]^T, scatter q/k/vT ----------------
__global__ __launch_bounds__(128) void gemm_qkv_kernel(const bf16_t* __restrict__ A,
                                                       const bf16_t* __restrict__ W,
                                                       bf16_t* __restrict__ q,
                                                       bf16_t* __restrict__ kk,
                                                       bf16_t* __restrict__ vT) {
  int lane = threadIdx.x & 31, wave = threadIdx.x >> 5;
  int hi = lane >> 4, lr = lane & 15;
  int m0 = (blockIdx.y * 4 + wave) * 32;
  int n0 = blockIdx.x * 64;
  const bf16_t* arow0 = A + (size_t)(m0 + lr) * Cc;
  const bf16_t* arow1 = A + (size_t)(m0 + 16 + lr) * Cc;
  const bf16_t* bp = W + (size_t)(n0 + lr) * Cc;
  f32x8 acc[2][4] = {};
  gemm_core_bf16(arow0, arow1, bp, bp + 16 * Cc, bp + 32 * Cc, bp + 48 * Cc, Cc, hi, acc);
#pragma unroll
  for (int t = 0; t < 4; ++t) {
    int o = n0 + t * 16 + lr;
    int sel = o / Cc; int rem = o - sel * Cc;
    int h = rem >> 6, d = rem & 63;
#pragma unroll
    for (int s = 0; s < 2; ++s) {
#pragma unroll
      for (int r = 0; r < 8; ++r) {
        int m = m0 + 16 * s + r + 8 * hi;
        int b_ = m >> 10, n = m & 1023;
        bf16_t val = (bf16_t)acc[s][t][r];
        size_t bh = (size_t)(b_ * Hh + h);
        if (sel == 0)      q [(bh * Nn + n) * Dd + d] = val;
        else if (sel == 1) kk[(bh * Nn + n) * Dd + d] = val;
        else               vT[(bh * Dd + d) * Nn + n] = val;
      }
    }
  }
}

// ---------------- attention: S = scale * q k^T, row softmax, fp32 -> d_out ----------------
__global__ __launch_bounds__(128) void attn_softmax_kernel(const bf16_t* __restrict__ q,
                                                           const bf16_t* __restrict__ kk,
                                                           float* __restrict__ P) {
  extern __shared__ float smem[];            // 4 * 16*1032 floats + 64 stats
  int lane = threadIdx.x & 31, wave = threadIdx.x >> 5;
  int hi = lane >> 4, lr = lane & 15;
  int idx = blockIdx.x * 4 + wave;
  int nt = idx & 63, bh = idx >> 6;
  int n0 = nt * 16;
  float* S = smem + (size_t)wave * 16 * 1032;
  float* stats = smem + 4 * 16 * 1032;
  const bf16_t* arow = q + ((size_t)bh * Nn + n0 + lr) * Dd;
  const bf16_t* krow = kk + ((size_t)bh * Nn + lr) * Dd;
  // A fragments are loop-invariant (K = D = 64): load once.
  bf16x16 a0 = load_frag_bf16(arow, hi);
  bf16x16 a1 = load_frag_bf16(arow + 32, hi);
  // ping-pong B fragments across the 64 key tiles (unrolled by 2, tail peeled)
  bf16x16 c0 = load_frag_bf16(krow, hi);
  bf16x16 c1 = load_frag_bf16(krow + 32, hi);
  bf16x16 d0, d1;
  for (int mt = 0; mt < 62; mt += 2) {
    const bf16_t* nb1 = krow + (size_t)(mt + 1) * 16 * Dd;
    d0 = load_frag_bf16(nb1, hi);
    d1 = load_frag_bf16(nb1 + 32, hi);
    {
      f32x8 acc = {};
      acc = wmma_bf16(a0, c0, acc);
      acc = wmma_bf16(a1, c1, acc);
#pragma unroll
      for (int r = 0; r < 8; ++r)
        S[(size_t)(r + 8 * hi) * 1032 + mt * 16 + lr] = acc[r] * SCALE;
    }
    const bf16_t* nb2 = krow + (size_t)(mt + 2) * 16 * Dd;
    c0 = load_frag_bf16(nb2, hi);
    c1 = load_frag_bf16(nb2 + 32, hi);
    {
      f32x8 acc = {};
      acc = wmma_bf16(a0, d0, acc);
      acc = wmma_bf16(a1, d1, acc);
#pragma unroll
      for (int r = 0; r < 8; ++r)
        S[(size_t)(r + 8 * hi) * 1032 + (mt + 1) * 16 + lr] = acc[r] * SCALE;
    }
  }
  {
    const bf16_t* nb = krow + (size_t)63 * 16 * Dd;
    d0 = load_frag_bf16(nb, hi);
    d1 = load_frag_bf16(nb + 32, hi);
    f32x8 acc = {};
    acc = wmma_bf16(a0, c0, acc);
    acc = wmma_bf16(a1, c1, acc);
#pragma unroll
    for (int r = 0; r < 8; ++r)
      S[(size_t)(r + 8 * hi) * 1032 + 62 * 16 + lr] = acc[r] * SCALE;
    f32x8 acc2 = {};
    acc2 = wmma_bf16(a0, d0, acc2);
    acc2 = wmma_bf16(a1, d1, acc2);
#pragma unroll
    for (int r = 0; r < 8; ++r)
      S[(size_t)(r + 8 * hi) * 1032 + 63 * 16 + lr] = acc2[r] * SCALE;
  }
  __syncthreads();
  // softmax: lane owns half a row (512 cols)
  float* row = S + (size_t)lr * 1032 + hi * 512;
  float mx = -3.0e38f;
  for (int i = 0; i < 512; ++i) mx = fmaxf(mx, row[i]);
  mx = fmaxf(mx, __shfl_xor(mx, 16, 32));
  float sum = 0.f;
  for (int i = 0; i < 512; ++i) { float e = __expf(row[i] - mx); row[i] = e; sum += e; }
  sum += __shfl_xor(sum, 16, 32);
  if (hi == 0) stats[wave * 16 + lr] = 1.0f / sum;
  __syncthreads();
  // coalesced normalized write
  float* Pout = P + ((size_t)bh * Nn + n0) * Nn;
  for (int r = 0; r < 16; ++r) {
    float inv = stats[wave * 16 + r];
    const float* srow = S + (size_t)r * 1032;
    for (int c0i = lane * 4; c0i < 1024; c0i += 128) {
      f32x4 vv = *(const f32x4*)(srow + c0i);
      vv = vv * inv;
      *(f32x4*)(Pout + (size_t)r * Nn + c0i) = vv;
    }
  }
}

// ---------------- in-place 1x1 head-mix conv + BatchNorm (eval) ----------------
__global__ __launch_bounds__(256) void mix_bn_kernel(float* __restrict__ attn,
                                                     const float* __restrict__ cw,
                                                     const float* __restrict__ cb,
                                                     const float* __restrict__ bng,
                                                     const float* __restrict__ bnb,
                                                     const float* __restrict__ rm,
                                                     const float* __restrict__ rv) {
  const size_t per_b = (size_t)Nn * Nn / 4;
  size_t tid = (size_t)blockIdx.x * 256 + threadIdx.x;
  if (tid >= (size_t)Bb * per_b) return;
  size_t b = tid / per_b, off = tid % per_b;
  f32x4* A4 = (f32x4*)attn;
  f32x4 p[Hh];
#pragma unroll
  for (int h = 0; h < Hh; ++h)
    p[h] = A4[(size_t)(b * Hh + h) * per_b + off];
  f32x4 o_[Hh];
#pragma unroll
  for (int o = 0; o < Hh; ++o) {
    float sc = bng[o] * rsqrtf(rv[o] + EPS);
    float bias = (cb[o] - rm[o]) * sc + bnb[o];
    f32x4 acc = {bias, bias, bias, bias};
#pragma unroll
    for (int h = 0; h < Hh; ++h) acc += p[h] * (cw[o * Hh + h] * sc);
    o_[o] = acc;
  }
#pragma unroll
  for (int o = 0; o < Hh; ++o)
    A4[(size_t)(b * Hh + o) * per_b + off] = o_[o];
}

// ---------------- attn @ V : fp32 attn (from d_out) x bf16 vT -> bf16 [B,N,C] ----------------
__global__ __launch_bounds__(128) void av_kernel(const float* __restrict__ attn,
                                                 const bf16_t* __restrict__ vT,
                                                 bf16_t* __restrict__ outv) {
  int lane = threadIdx.x & 31, wave = threadIdx.x >> 5;
  int hi = lane >> 4, lr = lane & 15;
  int idx = blockIdx.x * 4 + wave;
  int nt = idx & 31, bh = idx >> 5;          // 32-row tiles
  int b_ = bh / Hh, h = bh % Hh;
  int n0 = nt * 32;
  const float* arow0 = attn + ((size_t)bh * Nn + n0 + lr) * Nn;
  const float* arow1 = attn + ((size_t)bh * Nn + n0 + 16 + lr) * Nn;
  const bf16_t* bp = vT + ((size_t)bh * Dd + lr) * Nn;
  f32x8 acc[2][4] = {};
  gemm_core_f32a(arow0, arow1, bp, bp + 16 * Nn, bp + 32 * Nn, bp + 48 * Nn, Nn, hi, acc);
#pragma unroll
  for (int t = 0; t < 4; ++t) {
    int d = t * 16 + lr;
#pragma unroll
    for (int s = 0; s < 2; ++s) {
#pragma unroll
      for (int r = 0; r < 8; ++r) {
        int n = n0 + 16 * s + r + 8 * hi;
        outv[((size_t)(b_ * Nn + n)) * Cc + h * Dd + d] = (bf16_t)acc[s][t][r];
      }
    }
  }
}

// ---------------- proj GEMM + bias + residual -> fp32 x1 ----------------
__global__ __launch_bounds__(128) void gemm_proj_kernel(const bf16_t* __restrict__ A,
                                                        const bf16_t* __restrict__ W,
                                                        const float* __restrict__ bias,
                                                        const float* __restrict__ resid,
                                                        float* __restrict__ X1) {
  int lane = threadIdx.x & 31, wave = threadIdx.x >> 5;
  int hi = lane >> 4, lr = lane & 15;
  int m0 = (blockIdx.y * 4 + wave) * 32;
  int n0 = blockIdx.x * 64;
  const bf16_t* arow0 = A + (size_t)(m0 + lr) * Cc;
  const bf16_t* arow1 = A + (size_t)(m0 + 16 + lr) * Cc;
  const bf16_t* bp = W + (size_t)(n0 + lr) * Cc;
  f32x8 acc[2][4] = {};
  gemm_core_bf16(arow0, arow1, bp, bp + 16 * Cc, bp + 32 * Cc, bp + 48 * Cc, Cc, hi, acc);
#pragma unroll
  for (int t = 0; t < 4; ++t) {
    int col = n0 + t * 16 + lr;
#pragma unroll
    for (int s = 0; s < 2; ++s) {
#pragma unroll
      for (int r = 0; r < 8; ++r) {
        size_t m = (size_t)(m0 + 16 * s + r + 8 * hi);
        X1[m * Cc + col] = acc[s][t][r] + bias[col] + resid[m * Cc + col];
      }
    }
  }
}

// ---------------- fc1 GEMM + bias + exact GELU -> bf16 hid ----------------
__global__ __launch_bounds__(128) void gemm_fc1_kernel(const bf16_t* __restrict__ A,
                                                       const bf16_t* __restrict__ W,
                                                       const float* __restrict__ bias,
                                                       bf16_t* __restrict__ hid) {
  int lane = threadIdx.x & 31, wave = threadIdx.x >> 5;
  int hi = lane >> 4, lr = lane & 15;
  int m0 = (blockIdx.y * 4 + wave) * 32;
  int n0 = blockIdx.x * 64;
  const bf16_t* arow0 = A + (size_t)(m0 + lr) * Cc;
  const bf16_t* arow1 = A + (size_t)(m0 + 16 + lr) * Cc;
  const bf16_t* bp = W + (size_t)(n0 + lr) * Cc;
  f32x8 acc[2][4] = {};
  gemm_core_bf16(arow0, arow1, bp, bp + 16 * Cc, bp + 32 * Cc, bp + 48 * Cc, Cc, hi, acc);
#pragma unroll
  for (int t = 0; t < 4; ++t) {
    int col = n0 + t * 16 + lr;
#pragma unroll
    for (int s = 0; s < 2; ++s) {
#pragma unroll
      for (int r = 0; r < 8; ++r) {
        size_t m = (size_t)(m0 + 16 * s + r + 8 * hi);
        float xv = acc[s][t][r] + bias[col];
        float gel = 0.5f * xv * (1.0f + erff(xv * 0.70710678118654752f));
        hid[m * HID + col] = (bf16_t)gel;
      }
    }
  }
}

// ---------------- fc2 GEMM + bias + residual -> fp32 d_out x ----------------
__global__ __launch_bounds__(128) void gemm_fc2_kernel(const bf16_t* __restrict__ A,
                                                       const bf16_t* __restrict__ W,
                                                       const float* __restrict__ bias,
                                                       const float* __restrict__ X1,
                                                       float* __restrict__ outx) {
  int lane = threadIdx.x & 31, wave = threadIdx.x >> 5;
  int hi = lane >> 4, lr = lane & 15;
  int m0 = (blockIdx.y * 4 + wave) * 32;
  int n0 = blockIdx.x * 64;
  const bf16_t* arow0 = A + (size_t)(m0 + lr) * HID;
  const bf16_t* arow1 = A + (size_t)(m0 + 16 + lr) * HID;
  const bf16_t* bp = W + (size_t)(n0 + lr) * HID;
  f32x8 acc[2][4] = {};
  gemm_core_bf16(arow0, arow1, bp, bp + 16 * HID, bp + 32 * HID, bp + 48 * HID, HID, hi, acc);
#pragma unroll
  for (int t = 0; t < 4; ++t) {
    int col = n0 + t * 16 + lr;
#pragma unroll
    for (int s = 0; s < 2; ++s) {
#pragma unroll
      for (int r = 0; r < 8; ++r) {
        size_t m = (size_t)(m0 + 16 * s + r + 8 * hi);
        outx[m * Cc + col] = X1[m * Cc + col] + acc[s][t][r] + bias[col];
      }
    }
  }
}

// ---------------- host launcher ----------------
extern "C" void kernel_launch(void* const* d_in, const int* in_sizes, int n_in,
                              void* d_out, int out_size, void* d_ws, size_t ws_size,
                              hipStream_t stream) {
  const float* inputs = (const float*)d_in[0];
  const float* ln1_g  = (const float*)d_in[1];
  const float* ln1_b  = (const float*)d_in[2];
  const float* qkv_w  = (const float*)d_in[3];
  const float* conv_w = (const float*)d_in[4];
  const float* conv_b = (const float*)d_in[5];
  const float* bn_g   = (const float*)d_in[6];
  const float* bn_b   = (const float*)d_in[7];
  const float* bn_rm  = (const float*)d_in[8];
  const float* bn_rv  = (const float*)d_in[9];
  const float* proj_w = (const float*)d_in[10];
  const float* proj_b = (const float*)d_in[11];
  const float* ln2_g  = (const float*)d_in[12];
  const float* ln2_b  = (const float*)d_in[13];
  const float* fc1_w  = (const float*)d_in[14];
  const float* fc1_b  = (const float*)d_in[15];
  const float* fc2_w  = (const float*)d_in[16];
  const float* fc2_b  = (const float*)d_in[17];

  float* out_x    = (float*)d_out;
  float* out_attn = out_x + (size_t)Bb * Nn * Cc;

  char* ws = (char*)d_ws;
  bf16_t* xln  = (bf16_t*)(ws + OFF_XLN);
  bf16_t* qb   = (bf16_t*)(ws + OFF_Q);
  bf16_t* kb   = (bf16_t*)(ws + OFF_K);
  bf16_t* vtb  = (bf16_t*)(ws + OFF_VT);
  bf16_t* hid  = (bf16_t*)(ws + OFF_HID);   // aliases xln/q/k/vT (dead by then)
  bf16_t* avb  = (bf16_t*)(ws + OFF_AV);
  float*  x1   = (float*)(ws + OFF_X1);
  bf16_t* h2   = (bf16_t*)(ws + OFF_H2);
  bf16_t* wqkv = (bf16_t*)(ws + OFF_WQKV);
  bf16_t* wprj = (bf16_t*)(ws + OFF_WPRJ);
  bf16_t* wfc1 = (bf16_t*)(ws + OFF_WFC1);
  bf16_t* wfc2 = (bf16_t*)(ws + OFF_WFC2);

  auto blocks = [](int n) { return (n + 255) / 256; };

  // weights -> bf16
  cvt_kernel<<<blocks(3 * Cc * Cc), 256, 0, stream>>>(qkv_w, wqkv, 3 * Cc * Cc);
  cvt_kernel<<<blocks(Cc * Cc),     256, 0, stream>>>(proj_w, wprj, Cc * Cc);
  cvt_kernel<<<blocks(HID * Cc),    256, 0, stream>>>(fc1_w, wfc1, HID * Cc);
  cvt_kernel<<<blocks(Cc * HID),    256, 0, stream>>>(fc2_w, wfc2, Cc * HID);

  // LN1
  ln_kernel<<<Bb * Nn, 256, 0, stream>>>(inputs, ln1_g, ln1_b, xln);

  // QKV: M=8192 (32-row wave tiles), Nout=2304, K=768
  gemm_qkv_kernel<<<dim3(2304 / 64, 8192 / 128), 128, 0, stream>>>(xln, wqkv, qb, kb, vtb);

  // attention softmax -> d_out attn region (fp32)
  size_t shm = (size_t)(4 * 16 * 1032 + 64) * sizeof(float);   // ~258 KB, < 320 KB WGP LDS
  attn_softmax_kernel<<<(Bb * Hh * (Nn / 16)) / 4, 128, shm, stream>>>(qb, kb, out_attn);

  // in-place 12x12 head-mix + BN
  mix_bn_kernel<<<(int)(((size_t)Bb * Nn * Nn / 4 + 255) / 256), 256, 0, stream>>>(
      out_attn, conv_w, conv_b, bn_g, bn_b, bn_rm, bn_rv);

  // attn @ V  (reads fp32 attn from d_out, converts to bf16 on the fly)
  av_kernel<<<(Bb * Hh * (Nn / 32)) / 4, 128, 0, stream>>>(out_attn, vtb, avb);

  // proj + bias + residual -> x1 (fp32)
  gemm_proj_kernel<<<dim3(Cc / 64, 8192 / 128), 128, 0, stream>>>(avb, wprj, proj_b, inputs, x1);

  // LN2 -> h2 (bf16)
  ln_kernel<<<Bb * Nn, 256, 0, stream>>>(x1, ln2_g, ln2_b, h2);

  // fc1 + GELU -> hid (bf16)
  gemm_fc1_kernel<<<dim3(HID / 64, 8192 / 128), 128, 0, stream>>>(h2, wfc1, fc1_b, hid);

  // fc2 + bias + residual -> d_out x
  gemm_fc2_kernel<<<dim3(Cc / 64, 8192 / 128), 128, 0, stream>>>(hid, wfc2, fc2_b, x1, out_x);
}
